// NDDE_76888504533150
// MI455X (gfx1250) — matrix operand order repaired and checked by reading, not compiled, not run on hardware
//
#include <hip/hip_runtime.h>
#include <stdint.h>

// ---------------------------------------------------------------------------
// NDDE forward integrator for MI455X (gfx1250), wave32 + WMMA bf16.
//   traj[0] = x0_c; for s in 0..511:
//     y = traj[max(s-16,0)]
//     H = tanh([x|y] @ W1 + b1)          (GEMM1: M=64, K=1024, N=512)
//     traj[s+1] = x + dt*(H @ W2 + b2)   (GEMM2: M=64, K=512,  N=512)
// Persistent kernel: 16 WGs x 256 thr = 128 waves = 128 16x16 output tiles.
// ---------------------------------------------------------------------------

typedef __attribute__((ext_vector_type(16))) __bf16 v16bf;
typedef __attribute__((ext_vector_type(8)))  float  v8f;

#define B_   64
#define D_   512
#define KIN  1024
#define NBLK 16
#define TPB  256

__device__ __forceinline__ __bf16 f2bf(float f) {
  return (__bf16)f;   // fptrunc -> v_cvt_pk_bf16_f32 / native cvt (RNE)
}

// Branch-free tanh on the hardware transcendentals:
//   tanh(x) = sign(x) * (1 - e)/(1 + e),  e = exp(-2|x|) = exp2(-2|x|*log2e)
// |x| large -> e -> 0 -> +/-1 with no special-casing. One v_exp_f32 + one
// v_rcp_f32; accuracy ~1 ulp-ish, far below the bf16 matmul noise floor.
__device__ __forceinline__ float fast_tanh(float x) {
  float ax = __builtin_fabsf(x);
  float e  = __builtin_amdgcn_exp2f(ax * -2.885390081777927f);
  float t  = (1.0f - e) * __builtin_amdgcn_rcpf(1.0f + e);
  return __builtin_copysignf(t, x);
}

// Pack a row-major KxN fp32 weight into WMMA B-fragment order (bf16):
//   out[((nt*KB + kb)*32 + lane)*16 + e] = bf16( W[(kb*32 + (lane>>4)*8 + (e&7) + (e>>3)*16)*N + nt*16 + (lane&15)] )
// so each lane's 16 K-values for a (kb, n-tile) fragment are contiguous (32 B).
__global__ void pack_wmmaB(const float* __restrict__ W, __bf16* __restrict__ out,
                           int KB, int total) {
  int idx = blockIdx.x * blockDim.x + threadIdx.x;
  if (idx >= total) return;
  int e    = idx & 15;
  int lane = (idx >> 4) & 31;
  int rest = idx >> 9;
  int kb   = rest % KB;
  int nt   = rest / KB;
  int k = kb * 32 + (lane >> 4) * 8 + (e & 7) + (e >> 3) * 16;
  int n = nt * 16 + (lane & 15);
  out[idx] = f2bf(W[k * D_ + n]);
}

__global__ __launch_bounds__(TPB, 1)
void ndde_persistent(const float* __restrict__ x0_c,
                     const float* __restrict__ b1,
                     const float* __restrict__ b2,
                     const float* __restrict__ tau,
                     const int*   __restrict__ p_ntau,
                     const int*   __restrict__ p_nsteps,
                     const __bf16* __restrict__ pW1,
                     const __bf16* __restrict__ pW2,
                     __bf16* __restrict__ Hbf,
                     float* __restrict__ traj,
                     unsigned* __restrict__ ctr)
{
  const int tid  = threadIdx.x;
  const int lane = tid & 31;
  const int wave = tid >> 5;
  const int lrow = lane & 15;   // C/D: column within tile; A: row within tile
  const int hi   = lane >> 4;   // lane half selects K-subrange / M+8
  const int m0   = (wave & 3) * 16;                 // M-tile (0..3)
  const int nt   = blockIdx.x * 2 + (wave >> 2);    // N-tile (0..31)
  const int n0   = nt * 16;

  const int NT = p_ntau[0];
  const int NS = p_nsteps[0];
  const float dt = tau[0] / (float)NT;

  unsigned bar = 0;
  auto gbar = [&]() {
    __threadfence();
    __syncthreads();
    bar += NBLK;
    if (tid == 0) {
      atomicAdd(ctr, 1u);
      while (atomicAdd(ctr, 0u) < bar) __builtin_amdgcn_s_sleep(1);
    }
    __syncthreads();
    __threadfence();
  };

  // traj[0] = x0_c  (each block copies 4 rows)
  #pragma unroll
  for (int i = 0; i < (B_ * D_) / (NBLK * TPB); ++i) {
    int idx = blockIdx.x * ((B_ * D_) / NBLK) + i * TPB + tid;
    traj[idx] = x0_c[idx];
  }
  gbar();

  for (int s = 0; s < NS; ++s) {
    const int sy = (s >= NT) ? (s - NT) : 0;
    const float* Xrow = traj + (size_t)s  * (B_ * D_) + (m0 + lrow) * D_;
    const float* Yrow = traj + (size_t)sy * (B_ * D_) + (m0 + lrow) * D_;

    // ---- GEMM1: H = tanh(Z @ W1 + b1), Z = [X | Y], K = 1024 ----
    v8f acc = {0.f, 0.f, 0.f, 0.f, 0.f, 0.f, 0.f, 0.f};
    const __bf16* pB1 = pW1 + ((size_t)(nt * 32) * 32 + lane) * 16;
    for (int kb = 0; kb < 32; ++kb) {
      const int kbase = kb * 32 + hi * 8;
      // K 0..511 comes from X, 512..1023 from Y; runs never straddle 512.
      const float* Z = (kb < 16) ? (Xrow + kbase) : (Yrow + kbase - (KIN / 2));
      float4 p0 = *(const float4*)(Z);
      float4 p1 = *(const float4*)(Z + 4);
      float4 p2 = *(const float4*)(Z + 16);
      float4 p3 = *(const float4*)(Z + 20);
      v16bf a;
      a[0]  = f2bf(p0.x); a[1]  = f2bf(p0.y); a[2]  = f2bf(p0.z); a[3]  = f2bf(p0.w);
      a[4]  = f2bf(p1.x); a[5]  = f2bf(p1.y); a[6]  = f2bf(p1.z); a[7]  = f2bf(p1.w);
      a[8]  = f2bf(p2.x); a[9]  = f2bf(p2.y); a[10] = f2bf(p2.z); a[11] = f2bf(p2.w);
      a[12] = f2bf(p3.x); a[13] = f2bf(p3.y); a[14] = f2bf(p3.z); a[15] = f2bf(p3.w);
      v16bf b;
      __builtin_memcpy(&b, pB1 + (size_t)kb * 512, 32);
      acc = __builtin_amdgcn_wmma_f32_16x16x32_bf16(false, a, false, b,
                                                    (short)0, acc, false, false);
    }
    {
      const int n = n0 + lrow;
      const float bias = b1[n];
      #pragma unroll
      for (int j = 0; j < 8; ++j) {
        int m = m0 + hi * 8 + j;
        Hbf[m * D_ + n] = f2bf(fast_tanh(acc[j] + bias));
      }
    }
    gbar();

    // ---- GEMM2: traj[s+1] = X + dt*(H @ W2 + b2), K = 512 ----
    v8f acc2 = {0.f, 0.f, 0.f, 0.f, 0.f, 0.f, 0.f, 0.f};
    const __bf16* Hrow = Hbf + (m0 + lrow) * D_;
    const __bf16* pB2 = pW2 + ((size_t)(nt * 16) * 32 + lane) * 16;
    for (int kb = 0; kb < 16; ++kb) {
      const int kbase = kb * 32 + hi * 8;
      v16bf a;
      __builtin_memcpy(&a, Hrow + kbase, 16);                      // K runs +0..7
      __builtin_memcpy((char*)&a + 16, Hrow + kbase + 16, 16);     // K runs +16..23
      v16bf b;
      __builtin_memcpy(&b, pB2 + (size_t)kb * 512, 32);
      acc2 = __builtin_amdgcn_wmma_f32_16x16x32_bf16(false, a, false, b,
                                                     (short)0, acc2, false, false);
    }
    {
      const int n = n0 + lrow;
      const float bias = b2[n];
      const float* Xc = traj + (size_t)s * (B_ * D_);
      float* Xn = traj + (size_t)(s + 1) * (B_ * D_);
      #pragma unroll
      for (int j = 0; j < 8; ++j) {
        int m = m0 + hi * 8 + j;
        Xn[m * D_ + n] = Xc[m * D_ + n] + dt * (acc2[j] + bias);
      }
    }
    gbar();
  }
}

extern "C" void kernel_launch(void* const* d_in, const int* in_sizes, int n_in,
                              void* d_out, int out_size, void* d_ws, size_t ws_size,
                              hipStream_t stream) {
  (void)in_sizes; (void)n_in; (void)out_size; (void)ws_size;
  const float* x0   = (const float*)d_in[0];
  const float* W1   = (const float*)d_in[1];
  const float* b1   = (const float*)d_in[2];
  const float* W2   = (const float*)d_in[3];
  const float* b2   = (const float*)d_in[4];
  const float* tau  = (const float*)d_in[5];
  const int* ntau   = (const int*)d_in[6];
  const int* nsteps = (const int*)d_in[7];
  float* traj = (float*)d_out;

  char* ws = (char*)d_ws;
  unsigned* ctr = (unsigned*)ws;                              // 256 B
  __bf16*   Hbf = (__bf16*)(ws + 256);                        // 64 KB
  __bf16*   pW1 = (__bf16*)(ws + 256 + 65536);                // 1 MB
  __bf16*   pW2 = (__bf16*)(ws + 256 + 65536 + 1048576);      // 512 KB

  hipMemsetAsync(ctr, 0, 256, stream);
  pack_wmmaB<<<(KIN * D_ + 255) / 256, 256, 0, stream>>>(W1, pW1, 32, KIN * D_);
  pack_wmmaB<<<(D_ * D_ + 255) / 256, 256, 0, stream>>>(W2, pW2, 16, D_ * D_);
  ndde_persistent<<<NBLK, TPB, 0, stream>>>(x0, b1, b2, tau, ntau, nsteps,
                                            pW1, pW2, Hbf, traj, ctr);
}
